// DFConv2d_69020124446746
// MI455X (gfx1250) — matrix operand order repaired and checked by reading, not compiled, not run on hardware
//
#include <hip/hip_runtime.h>
#include <hip/hip_bf16.h>

typedef __attribute__((ext_vector_type(16))) _Float16 v16h;
typedef __attribute__((ext_vector_type(8)))  _Float16 v8h;
typedef __attribute__((ext_vector_type(8)))  float    v8f;

// Problem constants
// B=4, CIN=256, COUT=256, H=W=64, KS=3, PAD=1
// K (GEMM reduction) ordered as tap*256 + cin  -> 2304
// M = b*4096 + h*64 + w -> 16384 rows

// ---------------------------------------------------------------------------
// 1) Offset conv: om[b][oc][p] = conv3x3(x, w_off)[oc] + b_off[oc], oc in 0..26
// ---------------------------------------------------------------------------
__global__ __launch_bounds__(256)
void offset_conv_kernel(const float* __restrict__ x, const float* __restrict__ w_off,
                        const float* __restrict__ b_off, float* __restrict__ om)
{
    const int p  = blockIdx.x * 256 + threadIdx.x;   // 0..4095
    const int oc = blockIdx.y;                        // 0..26
    const int b  = blockIdx.z;                        // 0..3
    const int h  = p >> 6, w = p & 63;
    float acc = 0.0f;
    for (int c = 0; c < 256; ++c) {
        const float* xb = x + (((size_t)(b * 256 + c)) << 12);
        const float* wk = w_off + ((size_t)(oc * 256 + c)) * 9;
        #pragma unroll
        for (int ky = 0; ky < 3; ++ky) {
            const int yy = h + ky - 1;
            if (yy < 0 || yy > 63) continue;
            #pragma unroll
            for (int kx = 0; kx < 3; ++kx) {
                const int xx = w + kx - 1;
                if (xx < 0 || xx > 63) continue;
                acc += xb[yy * 64 + xx] * wk[ky * 3 + kx];
            }
        }
    }
    om[(((size_t)(b * 27 + oc)) << 12) + p] = acc + b_off[oc];
}

// ---------------------------------------------------------------------------
// 2) NCHW -> NHWC transpose of x (coalesced gathers for the sampler)
// ---------------------------------------------------------------------------
__global__ __launch_bounds__(256)
void nchw_to_nhwc_kernel(const float* __restrict__ x, float* __restrict__ xT)
{
    __shared__ float tile[32][33];
    const int b  = blockIdx.z;
    const int c0 = blockIdx.y * 32;
    const int p0 = blockIdx.x * 32;
    const int tx = threadIdx.x & 31;
    const int ty = threadIdx.x >> 5;   // 0..7
    #pragma unroll
    for (int i = 0; i < 4; ++i) {
        const int c = c0 + ty + i * 8;
        tile[ty + i * 8][tx] = x[(((size_t)(b * 256 + c)) << 12) + p0 + tx];
    }
    __syncthreads();
    #pragma unroll
    for (int i = 0; i < 4; ++i) {
        const int p = p0 + ty + i * 8;
        xT[((((size_t)b << 12) + p) << 8) + c0 + tx] = tile[tx][ty + i * 8];
    }
}

// ---------------------------------------------------------------------------
// 3) Pack conv weights to f16, K ordered as tap*256+cin: Wb[o][tap*256+cin]
// ---------------------------------------------------------------------------
__global__ __launch_bounds__(256)
void weight_pack_kernel(const float* __restrict__ w_conv, _Float16* __restrict__ Wb)
{
    const int t = blockIdx.x * 256 + threadIdx.x;
    if (t >= 256 * 2304) return;
    const int o   = t / 2304;
    const int k   = t - o * 2304;
    const int tap = k >> 8;
    const int c   = k & 255;
    Wb[t] = (_Float16)w_conv[((size_t)(o * 256 + c)) * 9 + tap];
}

// ---------------------------------------------------------------------------
// 4) Deformable bilinear sampling + mask -> im2col A[m][tap*256+cin]  (f16)
//    One block per spatial position m; 256 threads = 256 channels.
// ---------------------------------------------------------------------------
__global__ __launch_bounds__(256)
void im2col_sample_kernel(const float* __restrict__ xT, const float* __restrict__ om,
                          _Float16* __restrict__ A)
{
    __shared__ float so[27];
    const int m = blockIdx.x;            // 0..16383
    const int b = m >> 12;
    const int p = m & 4095;
    const int h = p >> 6, w = p & 63;
    const int c = threadIdx.x;           // channel
    if (c < 27) so[c] = om[(((size_t)(b * 27 + c)) << 12) + p];
    __syncthreads();

    const float* xb = xT + (((size_t)b << 12) << 8);   // b * 4096 * 256
    _Float16* arow = A + (size_t)m * 2304;

    #pragma unroll
    for (int tap = 0; tap < 9; ++tap) {
        const int ty = tap / 3, tx = tap % 3;
        const float dy = so[2 * tap];
        const float dx = so[2 * tap + 1];
        const float mk = 1.0f / (1.0f + __expf(-so[18 + tap]));
        const float py = (float)(h - 1 + ty) + dy;
        const float px = (float)(w - 1 + tx) + dx;
        const float y0f = floorf(py), x0f = floorf(px);
        const float wy = py - y0f, wx = px - x0f;
        const int y0 = (int)y0f, x0 = (int)x0f;
        float s = 0.0f;
        #pragma unroll
        for (int cy = 0; cy < 2; ++cy) {
            const int  yy = y0 + cy;
            const float fy = cy ? wy : (1.0f - wy);
            const bool vy = (yy >= 0) && (yy <= 63);
            const int  yc = min(max(yy, 0), 63);
            #pragma unroll
            for (int cx = 0; cx < 2; ++cx) {
                const int  xx = x0 + cx;
                const float fx = cx ? wx : (1.0f - wx);
                const bool v = vy && (xx >= 0) && (xx <= 63);
                const int  xc = min(max(xx, 0), 63);
                const float g = v ? xb[(((size_t)(yc * 64 + xc)) << 8) + c] : 0.0f;
                s += g * fy * fx;
            }
        }
        arow[tap * 256 + c] = (_Float16)(s * mk);
    }
}

// ---------------------------------------------------------------------------
// 5) GEMM via v_wmma_f32_16x16x32_f16:
//    C[m][n] = sum_k A[m][k] * Wb[n][k];  M=16384, N=256, K=2304
//    Block: 128x128 tile, BK=32, 8 wave32s; wave = 32x64 sub-tile (2x4 WMMA).
// ---------------------------------------------------------------------------
__global__ __launch_bounds__(256)
void gemm_wmma_kernel(const _Float16* __restrict__ A, const _Float16* __restrict__ Wb,
                      float* __restrict__ out)
{
    constexpr int BK = 32;
    __shared__ __align__(32) _Float16 As[128 * BK];
    __shared__ __align__(32) _Float16 Bs[128 * BK];

    const int m0   = blockIdx.x * 128;
    const int n0   = blockIdx.y * 128;
    const int tid  = threadIdx.x;
    const int wave = tid >> 5;          // 0..7
    const int lane = tid & 31;
    const int wr   = wave & 3;          // wave row  -> 32 rows
    const int wc   = wave >> 2;         // wave col  -> 64 cols
    const int half = lane >> 4;         // 0/1
    const int lm   = lane & 15;

    v8f acc[2][4] = {};

    for (int k0 = 0; k0 < 2304; k0 += BK) {
        __syncthreads();
        // Stage 128x32 f16 tiles of A and of Wb (as [n][k]) into LDS, 16B chunks.
        #pragma unroll
        for (int i = 0; i < 2; ++i) {
            const int ci  = tid + i * 256;          // 0..511
            const int row = ci >> 2;                // 0..127
            const int col = (ci & 3) << 3;          // 0,8,16,24 (elements)
            *(uint4*)(&As[row * BK + col]) =
                *(const uint4*)(&A[(size_t)(m0 + row) * 2304 + k0 + col]);
            *(uint4*)(&Bs[row * BK + col]) =
                *(const uint4*)(&Wb[(size_t)(n0 + row) * 2304 + k0 + col]);
        }
        if (k0 + BK < 2304) {
            __builtin_prefetch(&A [(size_t)(m0 + (tid >> 1)) * 2304 + k0 + BK], 0, 1);
            __builtin_prefetch(&Wb[(size_t)(n0 + (tid >> 1)) * 2304 + k0 + BK], 0, 1);
        }
        __syncthreads();

        // A fragments: lane holds row lm; 16 contiguous K at offset 16h
        v16h afrag[2];
        #pragma unroll
        for (int mt = 0; mt < 2; ++mt) {
            const _Float16* base = &As[(wr * 32 + mt * 16 + lm) * BK];
            union { v16h v; v8h h2[2]; } u;
            u.h2[0] = *(const v8h*)(base + 16 * half);
            u.h2[1] = *(const v8h*)(base + 16 * half + 8);
            afrag[mt] = u.v;
        }
        // B fragments: lane holds col lm; 16 contiguous K at offset 16h
        v16h bfrag[4];
        #pragma unroll
        for (int nt = 0; nt < 4; ++nt) {
            const _Float16* base = &Bs[(wc * 64 + nt * 16 + lm) * BK];
            bfrag[nt] = *(const v16h*)(base + 16 * half);
        }
        #pragma unroll
        for (int mt = 0; mt < 2; ++mt)
            #pragma unroll
            for (int nt = 0; nt < 4; ++nt)
                acc[mt][nt] = __builtin_amdgcn_wmma_f32_16x16x32_f16(
                    false, afrag[mt], false, bfrag[nt],
                    (short)0, acc[mt][nt], false, false);
    }

    // Store: C/D layout -> lane holds col lm, rows r + 8*half; out is NCHW.
    #pragma unroll
    for (int mt = 0; mt < 2; ++mt) {
        #pragma unroll
        for (int nt = 0; nt < 4; ++nt) {
            const int ng = n0 + wc * 64 + nt * 16 + lm;
            const int mb = m0 + wr * 32 + mt * 16 + half * 8;
            #pragma unroll
            for (int r = 0; r < 8; ++r) {
                const int mg = mb + r;
                const int bI = mg >> 12;
                const int pp = mg & 4095;
                out[(((size_t)(bI * 256 + ng)) << 12) + pp] = acc[mt][nt][r];
            }
        }
    }
}

// ---------------------------------------------------------------------------
extern "C" void kernel_launch(void* const* d_in, const int* in_sizes, int n_in,
                              void* d_out, int out_size, void* d_ws, size_t ws_size,
                              hipStream_t stream)
{
    (void)in_sizes; (void)n_in; (void)out_size; (void)ws_size;
    const float* x      = (const float*)d_in[0];   // (4,256,64,64)
    const float* w_off  = (const float*)d_in[1];   // (27,256,3,3)
    const float* b_off  = (const float*)d_in[2];   // (27,)
    const float* w_conv = (const float*)d_in[3];   // (256,256,3,3)
    float* out = (float*)d_out;                    // (4,256,64,64)

    char* ws = (char*)d_ws;
    size_t off = 0;
    auto carve = [&](size_t bytes) -> char* {
        char* p = ws + off;
        off = (off + bytes + 255) & ~(size_t)255;
        return p;
    };
    float*    om   = (float*)   carve((size_t)4 * 27 * 4096 * 4);     // 1.7 MB
    float*    xT   = (float*)   carve((size_t)4 * 4096 * 256 * 4);    // 16.8 MB
    _Float16* Wb   = (_Float16*)carve((size_t)256 * 2304 * 2);        // 1.2 MB
    _Float16* Acol = (_Float16*)carve((size_t)16384 * 2304 * 2);      // 75.5 MB

    offset_conv_kernel  <<<dim3(16, 27, 4), 256, 0, stream>>>(x, w_off, b_off, om);
    nchw_to_nhwc_kernel <<<dim3(128, 8, 4), 256, 0, stream>>>(x, xT);
    weight_pack_kernel  <<<dim3((256 * 2304 + 255) / 256), 256, 0, stream>>>(w_conv, Wb);
    im2col_sample_kernel<<<dim3(16384), 256, 0, stream>>>(xT, om, Acol);
    gemm_wmma_kernel    <<<dim3(128, 2), 256, 0, stream>>>(Acol, Wb, out);
}
